// Linear4Bit_47296179864027
// MI455X (gfx1250) — compile-verified, matrix operand order checked by reading, not run on hardware
//
#include <hip/hip_runtime.h>
#include <stdint.h>

// Problem is fixed by the reference: M = N = K = 4096, QUANT_BLOCK = 32.
#define MM 4096
#define NN 4096
#define KK 4096
#define QB 32          // K-step == quant block == WMMA K
#define TM 128         // C tile rows per block
#define TN 128         // C tile cols per block
#define LROW 34        // LDS row pitch in halves: 32 + 2 pad (68B = 17 banks, conflict-free)

typedef __attribute__((ext_vector_type(16))) _Float16 v16h;
typedef __attribute__((ext_vector_type(8)))  float    v8f;

union Frag { v16h v; uint32_t u[8]; };

// Pack two floats into two f16 halves of a dword (RNE via hardware convert).
__device__ __forceinline__ uint32_t pk2h(float a, float b) {
  union { _Float16 h[2]; uint32_t u; } r;
  r.h[0] = (_Float16)a;
  r.h[1] = (_Float16)b;
  return r.u;
}

__global__ __launch_bounds__(256) void w4a16_gemm(
    const float* __restrict__ X,    // [M,K] f32
    const int*   __restrict__ PW,   // [K/2,N] one byte (2 nibbles) per int32
    const float* __restrict__ S,    // [K/32,N]
    const float* __restrict__ Z,    // [K/32,N]
    const float* __restrict__ Bias, // [N]
    float*       __restrict__ C) {  // [M,N]
  __shared__ uint16_t Xs[TM * LROW]; // f16 bits of x tile, [m][k]
  __shared__ uint16_t Wt[TN * LROW]; // f16 bits of dequant W tile, TRANSPOSED: [n][k]

  const int t    = threadIdx.x;
  const int lane = t & 31;
  const int wave = t >> 5;
  const int m0 = blockIdx.y * TM;
  const int n0 = blockIdx.x * TN;
  const int wm = (wave & 3) * 32;   // wave row offset in tile (4 x 32 = 128)
  const int wn = (wave >> 2) * 64;  // wave col offset in tile (2 x 64 = 128)
  const int lh = lane >> 4;         // lane half (ISA A/B fragment layout)
  const int lm = lane & 15;

  // This thread's fixed 4-column group for the W-dequant stage.
  const int wc = (t & 31) << 2;     // 0..124
  const int wrow = t >> 5;          // 0..7 (kb row, +8 on second pass)

  v8f acc[2][4];
#pragma unroll
  for (int mi = 0; mi < 2; ++mi)
#pragma unroll
    for (int ni = 0; ni < 4; ++ni)
      acc[mi][ni] = (v8f){0.f, 0.f, 0.f, 0.f, 0.f, 0.f, 0.f, 0.f};

  for (int k0 = 0; k0 < KK; k0 += QB) {
    __syncthreads();  // previous compute phase done with LDS

    // ---- stage x tile: 128 rows x 32 k, f32 -> f16, float4 loads ----
#pragma unroll
    for (int i = 0; i < 4; ++i) {
      const int f = t + i * 256;        // 0..1023 float4 slots
      const int r = f >> 3;             // row 0..127
      const int c = (f & 7) << 2;       // k offset 0,4,..,28
      const float4 xv = *(const float4*)&X[(size_t)(m0 + r) * KK + k0 + c];
      *(uint32_t*)&Xs[r * LROW + c]     = pk2h(xv.x, xv.y);
      *(uint32_t*)&Xs[r * LROW + c + 2] = pk2h(xv.z, xv.w);
    }

    // ---- per-block scale/zero for this thread's 4 columns ----
    const int blk = k0 >> 5;
    float sc[4], zr[4];
#pragma unroll
    for (int j = 0; j < 4; ++j) {
      sc[j] = S[(size_t)blk * NN + n0 + wc + j];
      zr[j] = Z[(size_t)blk * NN + n0 + wc + j];
    }

    // ---- stage dequantized W tile, transposed into Wt[n][k] ----
    const int kb0 = k0 >> 1;            // packed row base (2 k per byte)
#pragma unroll
    for (int i = 0; i < 2; ++i) {
      const int r = wrow + i * 8;       // kb row in tile: 0..15
      const int4 pv = *(const int4*)&PW[(size_t)(kb0 + r) * NN + n0 + wc];
      const int* pp = (const int*)&pv;
#pragma unroll
      for (int j = 0; j < 4; ++j) {
        const int p = pp[j];
        const float w0 = sc[j] * ((float)(p & 15) - zr[j]);         // k = 2r   (low nibble)
        const float w1 = sc[j] * ((float)((p >> 4) & 15) - zr[j]);  // k = 2r+1 (high nibble)
        // adjacent k pair -> one aligned dword in the transposed tile
        *(uint32_t*)&Wt[(wc + j) * LROW + (r << 1)] = pk2h(w0, w1);
      }
    }

    // ---- prefetch next K-step's lines into cache (global_prefetch_b8) ----
    if (k0 + QB < KK) {
      const int r = t >> 3;
      const int c = (t & 7) << 2;
      __builtin_prefetch(&X[(size_t)(m0 + r) * KK + k0 + QB + c], 0, 0);
      __builtin_prefetch(&PW[(size_t)(kb0 + 16 + wrow) * NN + n0 + wc], 0, 0);
    }

    __syncthreads();  // tiles visible

    // ---- fragment gathers per ISA 16-bit A/B layout ----
    // VGPR v, lane-half h holds the f16 pair at K = ((v>>2)<<4)+((v&3)<<1)+(h<<3)
    Frag a[2], b[4];
#pragma unroll
    for (int mi = 0; mi < 2; ++mi) {
      const uint16_t* base = &Xs[(wm + mi * 16 + lm) * LROW];
#pragma unroll
      for (int v = 0; v < 8; ++v) {
        const int kb = ((v >> 2) << 4) + ((v & 3) << 1) + (lh << 3);
        a[mi].u[v] = *(const uint32_t*)&base[kb];
      }
    }
#pragma unroll
    for (int ni = 0; ni < 4; ++ni) {
      const uint16_t* base = &Wt[(wn + ni * 16 + lm) * LROW];
#pragma unroll
      for (int v = 0; v < 8; ++v) {
        const int kb = ((v >> 2) << 4) + ((v & 3) << 1) + (lh << 3);
        b[ni].u[v] = *(const uint32_t*)&base[kb];
      }
    }

    // ---- 8 WMMAs per wave per K-step ----
#pragma unroll
    for (int mi = 0; mi < 2; ++mi)
#pragma unroll
      for (int ni = 0; ni < 4; ++ni)
        acc[mi][ni] = __builtin_amdgcn_wmma_f32_16x16x32_f16(
            /*neg_a=*/false, a[mi].v, /*neg_b=*/false, b[ni].v,
            /*c_mod=*/(short)0, acc[mi][ni],
            /*reuse_a=*/false, /*reuse_b=*/false);
  }

  // ---- epilogue: bias add + store (C/D layout: VGPR r -> M = r + 8*lh, N = lm) ----
#pragma unroll
  for (int ni = 0; ni < 4; ++ni) {
    const int col = n0 + wn + ni * 16 + lm;
    const float bv = Bias[col];
#pragma unroll
    for (int mi = 0; mi < 2; ++mi) {
#pragma unroll
      for (int r = 0; r < 8; ++r) {
        const int row = m0 + wm + mi * 16 + lh * 8 + r;
        C[(size_t)row * NN + col] = acc[mi][ni][r] + bv;
      }
    }
  }
}

extern "C" void kernel_launch(void* const* d_in, const int* in_sizes, int n_in,
                              void* d_out, int out_size, void* d_ws, size_t ws_size,
                              hipStream_t stream) {
  const float* x  = (const float*)d_in[0];
  const int*   pw = (const int*)d_in[1];
  const float* s  = (const float*)d_in[2];
  const float* z  = (const float*)d_in[3];
  const float* b  = (const float*)d_in[4];
  float* out = (float*)d_out;

  dim3 grid(NN / TN, MM / TM);  // 32 x 32 tiles
  w4a16_gemm<<<grid, 256, 0, stream>>>(x, pw, s, z, b, out);
}
